// S_AlexCapsNet_Fully_CIFAR100_87488483820193
// MI455X (gfx1250) — matrix operand order repaired and checked by reading, not compiled, not run on hardware
//
#include <hip/hip_runtime.h>

// ---------------------------------------------------------------------------
// AlexCapsNet (CIFAR-100) forward on MI455X / gfx1250.
//  - conv / FC / caps-projection via v_wmma_f32_16x16x32_bf16 (fp32 accum)
//  - all weight B-operands pre-converted to bf16 in a K-chunked swizzle
//    Bsw[k/32][n][k%32] so async copies land directly in [n][k] LDS layout
//    (fragment reads stay vectorized ds_load_b128)
//  - tiles staged with global_load_async_to_lds_b128; GEMM double-buffers the
//    staging and overlaps it with WMMA via s_wait_asynccnt 0x3
//  - routing recomputes x_hat on the fly (avoids 944MB intermediate; Wcaps
//    stays resident in the 192MB L2)
// ---------------------------------------------------------------------------

typedef __attribute__((ext_vector_type(16))) __bf16 bf16x16;
typedef __attribute__((ext_vector_type(8)))  float  floatx8;

__device__ __forceinline__ void async_b128(unsigned lds_off, const void* gptr)
{
    asm volatile("global_load_async_to_lds_b128 %0, %1, off"
                 :: "v"(lds_off), "v"(gptr) : "memory");
}
__device__ __forceinline__ void async_b128_o16(unsigned lds_off, const void* gptr)
{
    asm volatile("global_load_async_to_lds_b128 %0, %1, off offset:16"
                 :: "v"(lds_off), "v"(gptr) : "memory");
}

// ======================= WMMA implicit-GEMM 3x3 conv =======================
// C[m,n], m = (b*Ho+yo)*Wo+xo, n = cout. Weights pre-swizzled bf16
// [Kpad/32][N][32] (zero-padded k rows). Block 256 thr = 8 waves, tile 64x64.
template<bool RELU>
__global__ __launch_bounds__(256)
void conv3x3_wmma(const float* __restrict__ in, const __bf16* __restrict__ wt,
                  const float* __restrict__ bias, float* __restrict__ out,
                  int Bn, int Cin, int Hin, int Win, int Cout,
                  int Ho, int Wo, int pad)
{
    const int K = Cin * 9;
    const int M = Bn * Ho * Wo;
    const int N = Cout;
    const int n0 = blockIdx.x * 64;
    const int m0 = blockIdx.y * 64;

    __shared__ __bf16 lds_a[64][40];   // [m][k] bf16, 80B row stride
    __shared__ __bf16 lds_b[64][40];   // [n][k] bf16, 80B row stride

    const int t    = threadIdx.x;
    const int lane = t & 31;
    const int wv   = t >> 5;           // wave 0..7 (wave32)
    const int wm   = (wv & 3) * 16;
    const int wn   = (wv >> 2) * 32;

    floatx8 acc0, acc1;
#pragma unroll
    for (int r = 0; r < 8; ++r) { acc0[r] = 0.f; acc1[r] = 0.f; }

    // A-loader mapping: thread t stages 8 consecutive k of row (t>>2)
    const int am = t >> 2;
    const int ak = (t & 3) * 8;
    const int gm = m0 + am;
    const bool vm = (gm < M);
    int ab = 0, ayo = 0, axo = 0;
    if (vm) {
        ab  = gm / (Ho * Wo);
        int rem = gm - ab * Ho * Wo;
        ayo = rem / Wo;
        axo = rem - ayo * Wo;
    }
    // B-loader mapping: thread t async-copies 8 bf16 of n-row (t>>2), swizzled
    // source: Bsw[(c*N + n0+n)*32 + kq]
    const int bn  = t >> 2;
    const int bkq = (t & 3) * 8;
    const __bf16* bptr = wt + ((size_t)(n0 + bn)) * 32 + bkq;
    const unsigned ldb = (unsigned)(size_t)&lds_b[bn][bkq];

    for (int k0 = 0; k0 < K; k0 += 32) {
        // ---- B tile: async global -> LDS (lands in [n][k] layout) ----
        async_b128(ldb, bptr);
        bptr += (size_t)N * 32;
        // ---- A tile: branchless im2col gather, fp32 -> bf16 ----
#pragma unroll
        for (int j = 0; j < 8; ++j) {
            int gk = k0 + ak + j;
            int ci = gk / 9;
            int r9 = gk - ci * 9;
            int ky = r9 / 3;
            int kx = r9 - ky * 3;
            int yi = ayo + ky - pad;
            int xi = axo + kx - pad;
            bool inb = vm & (gk < K) & (yi >= 0) & (yi < Hin) & (xi >= 0) & (xi < Win);
            size_t off = inb ? ((((size_t)ab * Cin + ci) * Hin + yi) * Win + xi)
                             : (size_t)0;
            float v = in[off];                 // unconditional, predicated addr
            lds_a[am][ak + j] = (__bf16)(inb ? v : 0.f);
        }
        asm volatile("s_wait_asynccnt 0x0" ::: "memory");
        __syncthreads();

        // ---- fragments in contiguous k order, k = 16*(lane/16) + s; all reads contiguous ----
        const int mrow = wm + (lane & 15);
        const int kb   = (lane >> 4) * 16;
        bf16x16 af;
#pragma unroll
        for (int j = 0; j < 8; ++j) af[j]     = lds_a[mrow][kb + j];
#pragma unroll
        for (int j = 0; j < 8; ++j) af[8 + j] = lds_a[mrow][kb + 8 + j];

        const int nrow = wn + (lane & 15);
        const int klo  = (lane >> 4) * 16;
        bf16x16 bf0, bf1;
#pragma unroll
        for (int j = 0; j < 16; ++j) bf0[j] = lds_b[nrow][klo + j];
#pragma unroll
        for (int j = 0; j < 16; ++j) bf1[j] = lds_b[nrow + 16][klo + j];

        acc0 = __builtin_amdgcn_wmma_f32_16x16x32_bf16(false, af, false, bf0,
                                                       (short)0, acc0, false, false);
        acc1 = __builtin_amdgcn_wmma_f32_16x16x32_bf16(false, af, false, bf1,
                                                       (short)0, acc1, false, false);
        __syncthreads();
    }

    // ---- epilogue: C layout = VGPR r: M = 8*(lane>=16)+r, N = lane&15 ----
    const int mbase = m0 + wm + (lane >> 4) * 8;
    const int nc0   = n0 + wn + (lane & 15);
    const int nc1   = nc0 + 16;
#pragma unroll
    for (int r = 0; r < 8; ++r) {
        int gmr = mbase + r;
        if (gmr < M) {
            int bo  = gmr / (Ho * Wo);
            int rem = gmr - bo * Ho * Wo;
            int yo  = rem / Wo;
            int xo  = rem - yo * Wo;
            if (nc0 < N) {
                float v = acc0[r] + bias[nc0];
                if (RELU) v = fmaxf(v, 0.f);
                out[(((size_t)bo * Cout + nc0) * Ho + yo) * Wo + xo] = v;
            }
            if (nc1 < N) {
                float v = acc1[r] + bias[nc1];
                if (RELU) v = fmaxf(v, 0.f);
                out[(((size_t)bo * Cout + nc1) * Ho + yo) * Wo + xo] = v;
            }
        }
    }
}

// ============ WMMA GEMM, double-buffered async staging, bf16 B =============
// A fp32 row-major [M,K]; B bf16 swizzled [K/32][N][32]; C = scale*A*B
// (+bias)(ReLU). REQUIRES: M covered exactly by 64-row blocks, K % 32 == 0.
// N ragged handled in epilogue; B buffer needs a small tail pad.
template<int ACT>
__global__ __launch_bounds__(256)
void gemm_wmma_async(const float* __restrict__ A, const __bf16* __restrict__ Bm,
                     const float* __restrict__ bias, float* __restrict__ C,
                     int M, int N, int K, float scale)
{
    const int n0 = blockIdx.x * 64;
    const int m0 = blockIdx.y * 64;

    __shared__ float  lds_a[2][64][36];   // fp32 A tiles (ping-pong)
    __shared__ __bf16 lds_b[2][64][40];   // bf16 B tiles [n][k] (ping-pong)

    const int t    = threadIdx.x;
    const int lane = t & 31;
    const int wv   = t >> 5;
    const int wm   = (wv & 3) * 16;
    const int wn   = (wv >> 2) * 32;

    floatx8 acc0, acc1;
#pragma unroll
    for (int r = 0; r < 8; ++r) { acc0[r] = 0.f; acc1[r] = 0.f; }

    // staging mapping: A -> thread t moves 8 floats (2 x b128) of row (t>>2);
    // B -> thread t moves 8 bf16 (1 x b128) of swizzled n-row (t>>2).
    const int am = t >> 2, aq = (t & 3) * 8;
    const int bn = t >> 2, bkq = (t & 3) * 8;
    const float*  aptr = A  + (size_t)(m0 + am) * K + aq;
    const __bf16* bptr = Bm + ((size_t)(n0 + bn)) * 32 + bkq;
    const unsigned lda0 = (unsigned)(size_t)&lds_a[0][am][aq];
    const unsigned lda1 = (unsigned)(size_t)&lds_a[1][am][aq];
    const unsigned ldb0 = (unsigned)(size_t)&lds_b[0][bn][bkq];
    const unsigned ldb1 = (unsigned)(size_t)&lds_b[1][bn][bkq];

    const int ntiles = K >> 5;

    // prologue: fill buffer 0
    async_b128(lda0, aptr);
    async_b128_o16(lda0, aptr);
    async_b128(ldb0, bptr);
    aptr += 32;
    bptr += (size_t)N * 32;

    for (int it = 0; it < ntiles; ++it) {
        const int cur = it & 1;
        if (it + 1 < ntiles) {
            // issue next tile into the other buffer, then wait only for the
            // current tile's 3 ops (in-order completion): <= 3 outstanding.
            async_b128(cur ? lda0 : lda1, aptr);
            async_b128_o16(cur ? lda0 : lda1, aptr);
            async_b128(cur ? ldb0 : ldb1, bptr);
            aptr += 32;
            bptr += (size_t)N * 32;
            asm volatile("s_wait_asynccnt 0x3" ::: "memory");
        } else {
            asm volatile("s_wait_asynccnt 0x0" ::: "memory");
        }
        __syncthreads();

        // ---- fragments in contiguous k order, k = 16*(lane/16) + s; all reads contiguous ----
        const int mrow = wm + (lane & 15);
        const int kb   = (lane >> 4) * 16;
        bf16x16 af;
#pragma unroll
        for (int j = 0; j < 8; ++j) af[j]     = (__bf16)lds_a[cur][mrow][kb + j];
#pragma unroll
        for (int j = 0; j < 8; ++j) af[8 + j] = (__bf16)lds_a[cur][mrow][kb + 8 + j];

        const int nrow = wn + (lane & 15);
        const int klo  = (lane >> 4) * 16;
        bf16x16 bf0, bf1;
#pragma unroll
        for (int j = 0; j < 16; ++j) bf0[j] = lds_b[cur][nrow][klo + j];
#pragma unroll
        for (int j = 0; j < 16; ++j) bf1[j] = lds_b[cur][nrow + 16][klo + j];

        acc0 = __builtin_amdgcn_wmma_f32_16x16x32_bf16(false, af, false, bf0,
                                                       (short)0, acc0, false, false);
        acc1 = __builtin_amdgcn_wmma_f32_16x16x32_bf16(false, af, false, bf1,
                                                       (short)0, acc1, false, false);
        __syncthreads();
    }

    const int mbase = m0 + wm + (lane >> 4) * 8;
    const int nc0   = n0 + wn + (lane & 15);
    const int nc1   = nc0 + 16;
#pragma unroll
    for (int r = 0; r < 8; ++r) {
        int gmr = mbase + r;
        if (nc0 < N) {
            float v = acc0[r] * scale;
            if (bias) v += bias[nc0];
            if (ACT == 1) v = fmaxf(v, 0.f);
            C[(size_t)gmr * N + nc0] = v;
        }
        if (nc1 < N) {
            float v = acc1[r] * scale;
            if (bias) v += bias[nc1];
            if (ACT == 1) v = fmaxf(v, 0.f);
            C[(size_t)gmr * N + nc1] = v;
        }
    }
}

// ================= weight pre-conversion (bf16 + swizzle) ==================
// All emit Bsw[k/32][n][k%32]: a 64-row async tile copy lands as [n][k] in LDS.

// row-major fp32 [K][N] -> swizzled bf16
__global__ void f32_to_bf16_sw(const float* __restrict__ in, __bf16* __restrict__ out,
                               int K, int N)
{
    size_t idx = (size_t)blockIdx.x * blockDim.x + threadIdx.x;
    if (idx >= (size_t)K * N) return;
    int k = (int)(idx / N);
    int n = (int)(idx - (size_t)k * N);
    out[(((size_t)(k >> 5)) * N + n) * 32 + (k & 31)] = (__bf16)in[idx];
}

// conv weights OIHW [Cout][K] -> swizzled bf16, rows K..Kpad-1 zeroed
__global__ void convw_to_bf16_sw(const float* __restrict__ w, __bf16* __restrict__ out,
                                 int Cout, int K, int Kpad)
{
    int idx = blockIdx.x * blockDim.x + threadIdx.x;
    int total = Kpad * Cout;
    if (idx >= total) return;
    int k = idx / Cout;
    int n = idx - k * Cout;
    float v = (k < K) ? w[(size_t)n * K + k] : 0.f;
    out[(((size_t)(k >> 5)) * Cout + n) * 32 + (k & 31)] = (__bf16)v;
}

// Wcaps[100,1152,16,8] -> swizzled bf16 with k=(i*8+e), n=(o*16+d), N=1600
__global__ void caps_to_bf16_sw(const float* __restrict__ W, __bf16* __restrict__ out)
{
    size_t idx = (size_t)blockIdx.x * blockDim.x + threadIdx.x;   // source index
    if (idx >= (size_t)100 * 1152 * 16 * 8) return;
    int e = idx & 7;
    int d = (idx >> 3) & 15;
    int i = (int)((idx >> 7) % 1152);
    int o = (int)(idx / ((size_t)1152 * 128));
    int k = i * 8 + e;
    int n = o * 16 + d;
    out[(((size_t)(k >> 5)) * 1600 + n) * 32 + (k & 31)] = (__bf16)W[idx];
}

// ============================ elementwise kernels ==========================

__global__ void maxpool3s2p1(const float* __restrict__ in, float* __restrict__ out,
                             int BC, int Hi, int Wi, int Ho, int Wo)
{
    int idx = blockIdx.x * blockDim.x + threadIdx.x;
    int total = BC * Ho * Wo;
    if (idx >= total) return;
    int xo = idx % Wo;
    int yo = (idx / Wo) % Ho;
    int bc = idx / (Wo * Ho);
    const float* p = in + (size_t)bc * Hi * Wi;
    float m = -INFINITY;
#pragma unroll
    for (int ky = 0; ky < 3; ++ky) {
        int yi = yo * 2 - 1 + ky;
        if (yi < 0 || yi >= Hi) continue;
#pragma unroll
        for (int kx = 0; kx < 3; ++kx) {
            int xi = xo * 2 - 1 + kx;
            if (xi < 0 || xi >= Wi) continue;
            m = fmaxf(m, p[yi * Wi + xi]);
        }
    }
    out[idx] = m;
}

// squash along last dim (dim = 8 or 16): s * (n2/(1+n2)) / sqrt(n2+eps)
__global__ void squash_vec(const float* __restrict__ in, float* __restrict__ out,
                           int nvec, int dim)
{
    int idx = blockIdx.x * blockDim.x + threadIdx.x;
    if (idx >= nvec) return;
    const float* p = in + (size_t)idx * dim;
    float* q = out + (size_t)idx * dim;
    float n2 = 0.f;
    for (int j = 0; j < dim; ++j) n2 += p[j] * p[j];
    float sc = (n2 / (1.f + n2)) * rsqrtf(n2 + 1e-8f);
    for (int j = 0; j < dim; ++j) q[j] = p[j] * sc;
}

// c[b,o,i] = softmax over o of bb[b,o,i]   (layout [b][100][1152])
__global__ void caps_softmax_o(const float* __restrict__ bb, float* __restrict__ c,
                               int total /* B*1152 */)
{
    int idx = blockIdx.x * blockDim.x + threadIdx.x;
    if (idx >= total) return;
    int b = idx / 1152;
    int i = idx - b * 1152;
    const float* p = bb + (size_t)b * 115200 + i;
    float mx = -1e30f;
    for (int o = 0; o < 100; ++o) mx = fmaxf(mx, p[(size_t)o * 1152]);
    float sum = 0.f;
    for (int o = 0; o < 100; ++o) sum += __expf(p[(size_t)o * 1152] - mx);
    float inv = 1.f / sum;
    float* q = c + (size_t)b * 115200 + i;
    for (int o = 0; o < 100; ++o)
        q[(size_t)o * 1152] = __expf(p[(size_t)o * 1152] - mx) * inv;
}

// s[b,o,d] = sum_i c[b,o,i] * x_hat[b,o,i,d],  x_hat recomputed from W,u.
// One block per (b,o); W tile streams from L2 (Wcaps resident, 59MB < 192MB).
__global__ __launch_bounds__(256)
void caps_route_s(const float* __restrict__ c, const float* __restrict__ u,
                  const float* __restrict__ W, float* __restrict__ s)
{
    int bo = blockIdx.x;              // b*100 + o
    int b = bo / 100;
    int o = bo - b * 100;
    int t = threadIdx.x;

    float acc[16];
#pragma unroll
    for (int d = 0; d < 16; ++d) acc[d] = 0.f;

    for (int i = t; i < 1152; i += 256) {
        float ci = c[(size_t)bo * 1152 + i];
        const float* up = u + ((size_t)b * 1152 + i) * 8;
        float ue[8];
#pragma unroll
        for (int e = 0; e < 8; ++e) ue[e] = up[e];
        const float* wp = W + ((size_t)o * 1152 + i) * 128;  // [16][8]
#pragma unroll
        for (int d = 0; d < 16; ++d) {
            float xh = 0.f;
#pragma unroll
            for (int e = 0; e < 8; ++e) xh += wp[d * 8 + e] * ue[e];
            acc[d] += ci * xh;
        }
    }

    __shared__ float red[256];
    for (int d = 0; d < 16; ++d) {
        red[t] = acc[d];
        __syncthreads();
        for (int sh = 128; sh > 0; sh >>= 1) {
            if (t < sh) red[t] += red[t + sh];
            __syncthreads();
        }
        if (t == 0) s[(size_t)bo * 16 + d] = red[0];
        __syncthreads();
    }
}

// bb[b,o,i] += sum_d x_hat[b,o,i,d] * v[b,o,d]
//            = sum_e u[b,i,e] * (sum_d W[o,i,d,e] * v[b,o,d])
__global__ void caps_bupdate(const float* __restrict__ u, const float* __restrict__ W,
                             const float* __restrict__ v, float* __restrict__ bb,
                             int total)
{
    int idx = blockIdx.x * blockDim.x + threadIdx.x;
    if (idx >= total) return;
    int i = idx % 1152;
    int o = (idx / 1152) % 100;
    int b = idx / 115200;
    const float* wp = W + ((size_t)o * 1152 + i) * 128;
    const float* vp = v + ((size_t)b * 100 + o) * 16;
    const float* up = u + ((size_t)b * 1152 + i) * 8;
    float vv[16];
#pragma unroll
    for (int d = 0; d < 16; ++d) vv[d] = vp[d];
    float delta = 0.f;
#pragma unroll
    for (int e = 0; e < 8; ++e) {
        float te = 0.f;
#pragma unroll
        for (int d = 0; d < 16; ++d) te += wp[d * 8 + e] * vv[d];
        delta += up[e] * te;
    }
    bb[idx] += delta;
}

// ================================ launcher =================================

extern "C" void kernel_launch(void* const* d_in, const int* in_sizes, int n_in,
                              void* d_out, int out_size, void* d_ws, size_t ws_size,
                              hipStream_t stream)
{
    const float* x     = (const float*)d_in[0];
    const float* cw1   = (const float*)d_in[1];
    const float* cb1   = (const float*)d_in[2];
    const float* cw2   = (const float*)d_in[3];
    const float* cb2   = (const float*)d_in[4];
    const float* cw3   = (const float*)d_in[5];
    const float* cb3   = (const float*)d_in[6];
    const float* pw    = (const float*)d_in[7];
    const float* pb    = (const float*)d_in[8];
    const float* Wcaps = (const float*)d_in[9];
    const float* fw1   = (const float*)d_in[10];
    const float* fb1   = (const float*)d_in[11];
    const float* fw2   = (const float*)d_in[12];
    const float* fb2   = (const float*)d_in[13];
    const float* fw3   = (const float*)d_in[14];
    const float* fb3   = (const float*)d_in[15];

    char* wsp = (char*)d_ws;
    auto carve = [&](size_t bytes) -> void* {
        void* p = (void*)wsp;
        wsp += ((bytes + 255) / 256) * 256;
        return p;
    };
    float* h1 = (float*)carve((size_t)128 * 96 * 32 * 32 * 4);   // conv1 out
    float* p1 = (float*)carve((size_t)128 * 96 * 16 * 16 * 4);   // pool1
    float* h2 = (float*)carve((size_t)128 * 256 * 16 * 16 * 4);  // conv2 out
    float* p2 = (float*)carve((size_t)128 * 256 * 8 * 8 * 4);    // pool2
    float* h3 = (float*)carve((size_t)128 * 384 * 8 * 8 * 4);    // conv3 out
    float* pc = (float*)carve((size_t)128 * 256 * 6 * 6 * 4);    // primarycaps conv
    float* u  = (float*)carve((size_t)128 * 1152 * 8 * 4);       // squashed caps
    float* bb = (float*)carve((size_t)128 * 100 * 1152 * 4);     // routing logits
    float* cc = (float*)carve((size_t)128 * 100 * 1152 * 4);     // coupling coeffs
    float* sb = (float*)carve((size_t)128 * 1600 * 4);           // s (pre-squash)
    float* vb = (float*)carve((size_t)128 * 1600 * 4);           // v (squashed)
    float* f1 = (float*)carve((size_t)128 * 4096 * 4);
    float* f2 = (float*)carve((size_t)128 * 4096 * 4);
    __bf16* cw1b = (__bf16*)carve((size_t)32 * 96 * 2);          // conv weights
    __bf16* cw2b = (__bf16*)carve((size_t)864 * 256 * 2);        //  bf16 swizzled
    __bf16* cw3b = (__bf16*)carve((size_t)2304 * 384 * 2);
    __bf16* pwb  = (__bf16*)carve((size_t)3456 * 256 * 2);
    __bf16* wcb  = (__bf16*)carve((size_t)9216 * 1600 * 2);      // Wcaps bf16 swizzled
    __bf16* fw1b = (__bf16*)carve((size_t)1600 * 4096 * 2);
    __bf16* fw2b = (__bf16*)carve((size_t)4096 * 4096 * 2);
    __bf16* fw3b = (__bf16*)carve((size_t)4096 * 100 * 2);
    (void)carve(4096);                                           // tail pad for async overreach

    // ---- weight pre-conversion to swizzled bf16 (B of all WMMA kernels) ----
    convw_to_bf16_sw<<<12, 256, 0, stream>>>(cw1, cw1b, 96, 27, 32);
    convw_to_bf16_sw<<<864, 256, 0, stream>>>(cw2, cw2b, 256, 864, 864);
    convw_to_bf16_sw<<<3456, 256, 0, stream>>>(cw3, cw3b, 384, 2304, 2304);
    convw_to_bf16_sw<<<3456, 256, 0, stream>>>(pw, pwb, 256, 3456, 3456);
    caps_to_bf16_sw<<<57600, 256, 0, stream>>>(Wcaps, wcb);
    f32_to_bf16_sw<<<25600, 256, 0, stream>>>(fw1, fw1b, 1600, 4096);
    f32_to_bf16_sw<<<65536, 256, 0, stream>>>(fw2, fw2b, 4096, 4096);
    f32_to_bf16_sw<<<1600, 256, 0, stream>>>(fw3, fw3b, 4096, 100);

    // ---- conv backbone ----
    conv3x3_wmma<true><<<dim3(2, 2048), 256, 0, stream>>>(
        x, cw1b, cb1, h1, 128, 3, 32, 32, 96, 32, 32, 1);
    maxpool3s2p1<<<12288, 256, 0, stream>>>(h1, p1, 128 * 96, 32, 32, 16, 16);
    conv3x3_wmma<true><<<dim3(4, 512), 256, 0, stream>>>(
        p1, cw2b, cb2, h2, 128, 96, 16, 16, 256, 16, 16, 1);
    maxpool3s2p1<<<8192, 256, 0, stream>>>(h2, p2, 128 * 256, 16, 16, 8, 8);
    conv3x3_wmma<true><<<dim3(6, 128), 256, 0, stream>>>(
        p2, cw3b, cb3, h3, 128, 256, 8, 8, 384, 8, 8, 1);
    conv3x3_wmma<false><<<dim3(4, 72), 256, 0, stream>>>(
        h3, pwb, pb, pc, 128, 384, 8, 8, 256, 6, 6, 0);

    // ---- primary caps squash: [128,1152,8] ----
    squash_vec<<<576, 256, 0, stream>>>(pc, u, 128 * 1152, 8);

    // ---- routing iter 0: c uniform => one big WMMA GEMM, scale 1/100 ----
    gemm_wmma_async<0><<<dim3(25, 2), 256, 0, stream>>>(
        u, wcb, nullptr, sb, 128, 1600, 9216, 0.01f);
    squash_vec<<<50, 256, 0, stream>>>(sb, vb, 128 * 100, 16);

    hipMemsetAsync(bb, 0, (size_t)128 * 115200 * sizeof(float), stream);
    caps_bupdate<<<57600, 256, 0, stream>>>(u, Wcaps, vb, bb, 128 * 115200);

    // ---- routing iter 1 ----
    caps_softmax_o<<<576, 256, 0, stream>>>(bb, cc, 128 * 1152);
    caps_route_s<<<12800, 256, 0, stream>>>(cc, u, Wcaps, sb);
    squash_vec<<<50, 256, 0, stream>>>(sb, vb, 128 * 100, 16);
    caps_bupdate<<<57600, 256, 0, stream>>>(u, Wcaps, vb, bb, 128 * 115200);

    // ---- routing iter 2 (final v) ----
    caps_softmax_o<<<576, 256, 0, stream>>>(bb, cc, 128 * 1152);
    caps_route_s<<<12800, 256, 0, stream>>>(cc, u, Wcaps, sb);
    squash_vec<<<50, 256, 0, stream>>>(sb, vb, 128 * 100, 16);   // vb == f [128,1600]

    // ---- FC head ----
    gemm_wmma_async<1><<<dim3(64, 2), 256, 0, stream>>>(
        vb, fw1b, fb1, f1, 128, 4096, 1600, 1.0f);
    gemm_wmma_async<1><<<dim3(64, 2), 256, 0, stream>>>(
        f1, fw2b, fb2, f2, 128, 4096, 4096, 1.0f);
    gemm_wmma_async<0><<<dim3(2, 2), 256, 0, stream>>>(
        f2, fw3b, fb3, (float*)d_out, 128, 100, 4096, 1.0f);
}
